// MapNet_18734647345156
// MI455X (gfx1250) — matrix operand order repaired and checked by reading, not compiled, not run on hardware
//
#include <hip/hip_runtime.h>
#include <stdint.h>

#define NN   100000   // nodes
#define EE   100000   // edges per scale per direction
#define ELR  30000    // left/right edges
#define NMC  128      // channels
#define NSC  6        // scales
#define NLY  4        // fusion layers

typedef __attribute__((ext_vector_type(16))) __bf16 v16bf;
typedef __attribute__((ext_vector_type(8)))  float  v8f;

union ABPack { uint4 u[2]; v16bf v; };

__device__ __forceinline__ uint16_t f32_to_bf16(float f) {
    uint32_t x = __float_as_uint(f);
    uint32_t r = x + 0x7FFFu + ((x >> 16) & 1u);   // round to nearest even
    return (uint16_t)(r >> 16);
}

// ---------------- f32 -> bf16 blob conversion ----------------
__global__ void cvt_bf16_kernel(const float* __restrict__ src,
                                uint16_t* __restrict__ dst, int count) {
    int i = blockIdx.x * blockDim.x + threadIdx.x;
    if (i < count) dst[i] = f32_to_bf16(src[i]);
}

// ---------------- stem: relu(x @ W1.T + b), W1 is [128,2] ----------------
__global__ void stem1_kernel(const float* __restrict__ xy,
                             const float* __restrict__ w,
                             const float* __restrict__ b,
                             uint16_t* __restrict__ out) {
    int idx = blockIdx.x * blockDim.x + threadIdx.x;
    if (idx >= NN * NMC) return;
    int n = idx >> 7, c = idx & (NMC - 1);
    float v = fmaf(w[2 * c], xy[2 * n], fmaf(w[2 * c + 1], xy[2 * n + 1], b[c]));
    out[idx] = f32_to_bf16(fmaxf(v, 0.0f));
}

// ---------------- dense node GEMM: Y[N,128] = X[N,128] @ W[128,128]^T ----------------
// One wave owns 16 rows and sweeps all 8 column tiles: A (4 K-chunks) is loaded
// once into registers, 32x v_wmma_f32_16x16x32_bf16 per wave.
__global__ __launch_bounds__(256) void gemm_node_kernel(
        const uint16_t* __restrict__ X, const uint16_t* __restrict__ W,
        float* __restrict__ Y, int nrows) {
    const int lane = threadIdx.x & 31;
    const int wave = threadIdx.x >> 5;
    const int l16  = lane & 15;
    const int half = lane >> 4;
    const int rowbase = (blockIdx.x * 8 + wave) * 16;
    if (rowbase >= nrows) return;          // uniform per-wave exit, EXEC stays full

    const uint16_t* xrow = X + (size_t)(rowbase + l16) * NMC;
    ABPack aP[4];
    #pragma unroll
    for (int c = 0; c < 4; ++c) {
        aP[c].u[0] = *(const uint4*)(xrow + c * 32 + half * 8);        // K half*8..+7
        aP[c].u[1] = *(const uint4*)(xrow + c * 32 + 16 + half * 8);   // K 16+half*8..+7
    }

    #pragma unroll 2
    for (int t = 0; t < 8; ++t) {
        const int colbase = t * 16;
        const uint16_t* wrow = W + (size_t)(colbase + l16) * NMC;
        v8f acc = {};
        #pragma unroll
        for (int c = 0; c < 4; ++c) {
            ABPack bm;
            bm.u[0] = *(const uint4*)(wrow + c * 32 + half * 8);
            bm.u[1] = *(const uint4*)(wrow + c * 32 + 16 + half * 8);
            acc = __builtin_amdgcn_wmma_f32_16x16x32_bf16(
                    false, aP[c].v, false, bm.v, (short)0, acc, false, false);
        }
        #pragma unroll
        for (int r = 0; r < 8; ++r) {
            int m = rowbase + r + half * 8;                  // C/D layout: VGPR r
            Y[(size_t)m * NMC + colbase + l16] = acc[r];
        }
    }
}

// ---------------- edge gather->GEMM->scatter-add ----------------
// temp[u[e]] += X[v[e]] @ W^T ; one wave owns 16 edges, sweeps all 8 col tiles,
// gathered A tile loaded once, f32 atomic scatter.
__global__ __launch_bounds__(256) void gemm_edge_kernel(
        const uint16_t* __restrict__ X,
        const int* __restrict__ eu, const int* __restrict__ ev,
        const uint16_t* __restrict__ W, float* __restrict__ T, int ne) {
    const int lane = threadIdx.x & 31;
    const int wave = threadIdx.x >> 5;
    const int l16  = lane & 15;
    const int half = lane >> 4;
    const int ebase = (blockIdx.x * 8 + wave) * 16;
    if (ebase >= ne) return;               // ne % 16 == 0: active waves are full

    const uint16_t* xrow = X + (size_t)ev[ebase + l16] * NMC;   // gather A row
    ABPack aP[4];
    #pragma unroll
    for (int c = 0; c < 4; ++c) {
        aP[c].u[0] = *(const uint4*)(xrow + c * 32 + half * 8);
        aP[c].u[1] = *(const uint4*)(xrow + c * 32 + 16 + half * 8);
    }
    int ru[8];
    #pragma unroll
    for (int r = 0; r < 8; ++r) ru[r] = eu[ebase + r + half * 8];

    #pragma unroll 2
    for (int t = 0; t < 8; ++t) {
        const int colbase = t * 16;
        const uint16_t* wrow = W + (size_t)(colbase + l16) * NMC;
        v8f acc = {};
        #pragma unroll
        for (int c = 0; c < 4; ++c) {
            ABPack bm;
            bm.u[0] = *(const uint4*)(wrow + c * 32 + half * 8);
            bm.u[1] = *(const uint4*)(wrow + c * 32 + 16 + half * 8);
            acc = __builtin_amdgcn_wmma_f32_16x16x32_bf16(
                    false, aP[c].v, false, bm.v, (short)0, acc, false, false);
        }
        #pragma unroll
        for (int r = 0; r < 8; ++r)
            atomicAdd(&T[(size_t)ru[r] * NMC + colbase + l16], acc[r]);
    }
}

// ---------------- GroupNorm(1 group) epilogue: one wave per node ----------------
__global__ __launch_bounds__(256) void gn_kernel(
        const float* __restrict__ in, const float* __restrict__ gw,
        const float* __restrict__ gb, const float* __restrict__ addsrc,
        uint16_t* __restrict__ outbf, float* __restrict__ outf, int do_relu) {
    const int lane = threadIdx.x & 31;
    const int wave = threadIdx.x >> 5;
    const int node = blockIdx.x * 8 + wave;
    if (node >= NN) return;                // node uniform per wave

    const size_t base = (size_t)node * NMC + lane * 4;
    const float4 x = *(const float4*)(in + base);
    float s  = x.x + x.y + x.z + x.w;
    float s2 = x.x * x.x + x.y * x.y + x.z * x.z + x.w * x.w;
    #pragma unroll
    for (int off = 16; off > 0; off >>= 1) {
        s  += __shfl_xor(s,  off, 32);
        s2 += __shfl_xor(s2, off, 32);
    }
    const float mu  = s * (1.0f / NMC);
    const float var = s2 * (1.0f / NMC) - mu * mu;
    const float rs  = rsqrtf(var + 1e-5f);

    const float4 w4 = *(const float4*)(gw + lane * 4);
    const float4 b4 = *(const float4*)(gb + lane * 4);
    float4 r;
    r.x = (x.x - mu) * rs * w4.x + b4.x;
    r.y = (x.y - mu) * rs * w4.y + b4.y;
    r.z = (x.z - mu) * rs * w4.z + b4.z;
    r.w = (x.w - mu) * rs * w4.w + b4.w;
    if (addsrc) {
        const float4 a4 = *(const float4*)(addsrc + base);
        r.x += a4.x; r.y += a4.y; r.z += a4.z; r.w += a4.w;
    }
    if (do_relu) {
        r.x = fmaxf(r.x, 0.0f); r.y = fmaxf(r.y, 0.0f);
        r.z = fmaxf(r.z, 0.0f); r.w = fmaxf(r.w, 0.0f);
    }
    if (outf) *(float4*)(outf + base) = r;
    if (outbf) {
        uint2 p;
        p.x = (uint32_t)f32_to_bf16(r.x) | ((uint32_t)f32_to_bf16(r.y) << 16);
        p.y = (uint32_t)f32_to_bf16(r.z) | ((uint32_t)f32_to_bf16(r.w) << 16);
        *(uint2*)(outbf + base) = p;
    }
}

// =====================================================================
extern "C" void kernel_launch(void* const* d_in, const int* in_sizes, int n_in,
                              void* d_out, int out_size, void* d_ws, size_t ws_size,
                              hipStream_t stream) {
    (void)in_sizes; (void)n_in; (void)out_size; (void)ws_size;
    const float* ctrs    = (const float*)d_in[0];
    const float* feats   = (const float*)d_in[1];
    const int*   pre_u   = (const int*)d_in[2];
    const int*   pre_v   = (const int*)d_in[3];
    const int*   suc_u   = (const int*)d_in[4];
    const int*   suc_v   = (const int*)d_in[5];
    const int*   left_u  = (const int*)d_in[6];
    const int*   left_v  = (const int*)d_in[7];
    const int*   right_u = (const int*)d_in[8];
    const int*   right_v = (const int*)d_in[9];
    const float* in1_w   = (const float*)d_in[10];
    const float* in1_b   = (const float*)d_in[11];
    const float* in2_w   = (const float*)d_in[12];
    const float* in_gn_w = (const float*)d_in[13];
    const float* in_gn_b = (const float*)d_in[14];
    const float* seg1_w  = (const float*)d_in[15];
    const float* seg1_b  = (const float*)d_in[16];
    const float* seg2_w  = (const float*)d_in[17];
    const float* seg_gn_w= (const float*)d_in[18];
    const float* seg_gn_b= (const float*)d_in[19];
    const float* ctr_w   = (const float*)d_in[20];
    const float* pre_w   = (const float*)d_in[21];
    const float* suc_w   = (const float*)d_in[22];
    const float* left_w  = (const float*)d_in[23];
    const float* right_w = (const float*)d_in[24];
    const float* norm_w  = (const float*)d_in[25];
    const float* norm_b  = (const float*)d_in[26];
    const float* ctr2_w  = (const float*)d_in[27];
    const float* ctr2_gn_w = (const float*)d_in[28];
    const float* ctr2_gn_b = (const float*)d_in[29];

    // -------- workspace partition --------
    char* ws = (char*)d_ws;
    size_t off = 0;
    float* temp  = (float*)(ws + off); off += (size_t)NN * NMC * sizeof(float);
    float* res   = (float*)(ws + off); off += (size_t)NN * NMC * sizeof(float);
    uint16_t* featbf = (uint16_t*)(ws + off); off += (size_t)NN * NMC * sizeof(uint16_t);
    const size_t WSZ = (size_t)NMC * NMC;   // 16384 elems per 128x128 weight
    uint16_t* bf_in2   = (uint16_t*)(ws + off); off += WSZ * 2;
    uint16_t* bf_seg2  = (uint16_t*)(ws + off); off += WSZ * 2;
    uint16_t* bf_ctr   = (uint16_t*)(ws + off); off += NLY * WSZ * 2;
    uint16_t* bf_pre   = (uint16_t*)(ws + off); off += (size_t)NLY * NSC * WSZ * 2;
    uint16_t* bf_suc   = (uint16_t*)(ws + off); off += (size_t)NLY * NSC * WSZ * 2;
    uint16_t* bf_left  = (uint16_t*)(ws + off); off += NLY * WSZ * 2;
    uint16_t* bf_right = (uint16_t*)(ws + off); off += NLY * WSZ * 2;
    uint16_t* bf_ctr2  = (uint16_t*)(ws + off); off += NLY * WSZ * 2;

    // -------- convert weights to bf16 --------
    auto cvt = [&](const float* src, uint16_t* dst, int cnt) {
        cvt_bf16_kernel<<<(cnt + 255) / 256, 256, 0, stream>>>(src, dst, cnt);
    };
    cvt(in2_w,  bf_in2,  (int)WSZ);
    cvt(seg2_w, bf_seg2, (int)WSZ);
    cvt(ctr_w,  bf_ctr,  (int)(NLY * WSZ));
    cvt(pre_w,  bf_pre,  (int)(NLY * NSC * WSZ));
    cvt(suc_w,  bf_suc,  (int)(NLY * NSC * WSZ));
    cvt(left_w, bf_left, (int)(NLY * WSZ));
    cvt(right_w,bf_right,(int)(NLY * WSZ));
    cvt(ctr2_w, bf_ctr2, (int)(NLY * WSZ));

    const int GN_GRID   = NN / 8;              // 12500 blocks, 8 nodes (waves) each
    const int GEMM_GRID = (NN + 127) / 128;    // 782 blocks, 16 rows per wave
    const int EDGE_GRID = (EE + 127) / 128;    // 782
    const int SIDE_GRID = (ELR + 127) / 128;   // 235

    // -------- input stem: fi = GN(relu(ctrs@in1.T+b) @ in2.T) --------
    stem1_kernel<<<(NN * NMC + 255) / 256, 256, 0, stream>>>(ctrs, in1_w, in1_b, featbf);
    gemm_node_kernel<<<GEMM_GRID, 256, 0, stream>>>(featbf, bf_in2, temp, NN);
    gn_kernel<<<GN_GRID, 256, 0, stream>>>(temp, in_gn_w, in_gn_b,
                                           nullptr, nullptr, res, 0);   // res := fi
    // -------- seg stem + combine: feat = relu(fi + GN(...)) --------
    stem1_kernel<<<(NN * NMC + 255) / 256, 256, 0, stream>>>(feats, seg1_w, seg1_b, featbf);
    gemm_node_kernel<<<GEMM_GRID, 256, 0, stream>>>(featbf, bf_seg2, temp, NN);
    gn_kernel<<<GN_GRID, 256, 0, stream>>>(temp, seg_gn_w, seg_gn_b,
                                           res, featbf, res, 1);        // feat, res := relu(fi+fs)

    // -------- 4 fusion layers --------
    for (int i = 0; i < NLY; ++i) {
        // temp = feat @ ctr_w[i].T
        gemm_node_kernel<<<GEMM_GRID, 256, 0, stream>>>(featbf, bf_ctr + i * WSZ, temp, NN);
        // multi-scale pre/suc scatter-adds
        for (int k = 0; k < NSC; ++k) {
            gemm_edge_kernel<<<EDGE_GRID, 256, 0, stream>>>(
                featbf, pre_u + (size_t)k * EE, pre_v + (size_t)k * EE,
                bf_pre + ((size_t)i * NSC + k) * WSZ, temp, EE);
            gemm_edge_kernel<<<EDGE_GRID, 256, 0, stream>>>(
                featbf, suc_u + (size_t)k * EE, suc_v + (size_t)k * EE,
                bf_suc + ((size_t)i * NSC + k) * WSZ, temp, EE);
        }
        gemm_edge_kernel<<<SIDE_GRID, 256, 0, stream>>>(
            featbf, left_u, left_v, bf_left + i * WSZ, temp, ELR);
        gemm_edge_kernel<<<SIDE_GRID, 256, 0, stream>>>(
            featbf, right_u, right_v, bf_right + i * WSZ, temp, ELR);
        // feat = relu(GN(temp))
        gn_kernel<<<GN_GRID, 256, 0, stream>>>(temp, norm_w + i * NMC, norm_b + i * NMC,
                                               nullptr, featbf, nullptr, 1);
        // y2 = feat @ ctr2_w[i].T
        gemm_node_kernel<<<GEMM_GRID, 256, 0, stream>>>(featbf, bf_ctr2 + i * WSZ, temp, NN);
        // feat = relu(GN(y2) + res); res = feat
        gn_kernel<<<GN_GRID, 256, 0, stream>>>(temp, ctr2_gn_w + i * NMC, ctr2_gn_b + i * NMC,
                                               res, featbf, res, 1);
    }

    // final feat (f32) lives in res
    hipMemcpyAsync(d_out, res, (size_t)NN * NMC * sizeof(float),
                   hipMemcpyDeviceToDevice, stream);
}
